// MultiHeadAttention_8924942041293
// MI455X (gfx1250) — matrix operand order repaired and checked by reading, not compiled, not run on hardware
//
#include <hip/hip_runtime.h>
#include <hip/hip_bf16.h>

typedef __attribute__((ext_vector_type(16))) _Float16 v16h;
typedef __attribute__((ext_vector_type(8)))  float    v8f;
typedef __attribute__((ext_vector_type(4)))  int      v4i;

#define WMMA_F32_F16(a, b, c) \
    __builtin_amdgcn_wmma_f32_16x16x32_f16(false, (a), false, (b), (short)0, (c), false, false)

#if defined(__has_builtin)
#if __has_builtin(__builtin_amdgcn_global_load_async_to_lds_b128)
#define HAVE_ASYNC_LDS 1
#endif
#endif
#ifndef HAVE_ASYNC_LDS
#define HAVE_ASYNC_LDS 0
#endif

static constexpr int Bb  = 2;
static constexpr int Ss  = 4096;
static constexpr int Dd  = 1024;
static constexpr int Hh  = 16;
static constexpr int DHh = 64;
static constexpr int BS  = Bb * Ss;   // 8192 rows
static constexpr int LDSP = 40;       // padded LDS row (halves): 80B stride, multiple of 16B

union Frag16 { v16h h; uint4 u[2]; };

// A-matrix 16x32 f16 fragment (ISA 7.12.2): lane<16 -> row=lane, K = {k0..k0+7, k0+16..k0+23}
//                                           lane>=16 -> row=lane-16, K = {k0+8..k0+15, k0+24..k0+31}
static __device__ __forceinline__ v16h load_frag_a(const _Float16* row, int k0, int lane) {
    const _Float16* base = row + k0 + ((lane < 16) ? 0 : 8);
    Frag16 f;
    f.u[0] = *reinterpret_cast<const uint4*>(base);
    f.u[1] = *reinterpret_cast<const uint4*>(base + 16);
    return f.h;
}

// B-matrix 32x16 f16 fragment: lane = column; lane<16 -> K = {k0..k0+7, k0+16..k0+23}, lane>=16 -> K = {k0+8..k0+15, k0+24..k0+31}.
// `col` points at the column's contiguous K-major storage.
static __device__ __forceinline__ v16h load_frag_b(const _Float16* col, int k0, int lane) {
    const _Float16* base = col + k0 + ((lane < 16) ? 0 : 8);
    Frag16 f;
    f.u[0] = *reinterpret_cast<const uint4*>(base);
    f.u[1] = *reinterpret_cast<const uint4*>(base + 16);
    return f.h;
}

// Async global->LDS 16-byte copy (CDNA5 GLOBAL_LOAD_ASYNC_TO_LDS_B128, ASYNCcnt-tracked)
#if HAVE_ASYNC_LDS
static __device__ __forceinline__ void async_cp16(const _Float16* g, _Float16* l) {
    __builtin_amdgcn_global_load_async_to_lds_b128(
        (v4i*)g,
        (__attribute__((address_space(3))) v4i*)l,
        0, 0);
}
#endif

// ---------------- conversion kernels ----------------

__global__ void cvt_f32_to_f16(const float* __restrict__ in, _Float16* __restrict__ out, int n) {
    int i = blockIdx.x * blockDim.x + threadIdx.x;
    if (i < n) out[i] = (_Float16)in[i];
}

// Wt[n][k] = W[k][n]  (store weights column-major so B-fragments are contiguous over K)
__global__ void transpose_w_f16(const float* __restrict__ W, _Float16* __restrict__ Wt) {
    int i = blockIdx.x * blockDim.x + threadIdx.x;
    if (i < Dd * Dd) {
        int n = i >> 10;        // i / Dd
        int k = i & (Dd - 1);   // i % Dd
        Wt[(size_t)n * Dd + k] = (_Float16)W[(size_t)k * Dd + n];
    }
}

// ---------------- GEMM core: LDS-staged, double-buffered B tile ----------------
// Block = 256 threads (8 waves). Block tile: 128 rows x 64 cols. K step = 32.
// B tile (64 cols x 32 K halves) staged cooperatively into LDS (async when available),
// so the 8 waves share one copy instead of 8 redundant global fetches.
#define GEMM_ACC_BODY(A_, Bt_, ldsB_, acc_)                                            \
    const int lane = threadIdx.x & 31;                                                 \
    const int w    = threadIdx.x >> 5;                                                 \
    const int m0   = blockIdx.y * 128 + w * 16;                                        \
    const int n0   = blockIdx.x * 64;                                                  \
    const int nn   = lane & 15;                                                        \
    const int tcol  = threadIdx.x >> 2;        /* 0..63: staged column            */   \
    const int tpart = (threadIdx.x & 3) * 8;   /* 0,8,16,24 halves (16B chunks)   */   \
    const _Float16* arow = (A_) + (size_t)(m0 + nn) * Dd;                              \
    const _Float16* gB   = (Bt_) + (size_t)(n0 + tcol) * Dd + tpart;                   \
    {                                                                                  \
        const v8f zero = {0.f, 0.f, 0.f, 0.f, 0.f, 0.f, 0.f, 0.f};                     \
        _Pragma("unroll") for (int t = 0; t < 4; ++t) (acc_)[t] = zero;                \
    }

#if HAVE_ASYNC_LDS
#define GEMM_K_LOOP(ldsB_, acc_)                                                       \
    async_cp16(gB, &(ldsB_)[0][tcol][tpart]);                                          \
    for (int kt = 0; kt < Dd / 32; ++kt) {                                             \
        asm volatile("s_wait_asynccnt 0" ::: "memory");                                \
        __syncthreads();                                                               \
        if (kt + 1 < Dd / 32)                                                          \
            async_cp16(gB + (kt + 1) * 32, &(ldsB_)[(kt + 1) & 1][tcol][tpart]);       \
        const int k0  = kt * 32;                                                       \
        const int buf = kt & 1;                                                        \
        if (k0 + 32 < Dd) __builtin_prefetch(arow + k0 + 32, 0, 3);                    \
        v16h a = load_frag_a(arow, k0, lane);                                          \
        _Pragma("unroll") for (int t = 0; t < 4; ++t) {                                \
            v16h b = load_frag_b(&(ldsB_)[buf][t * 16 + nn][0], 0, lane);              \
            (acc_)[t] = WMMA_F32_F16(a, b, (acc_)[t]);                                 \
        }                                                                              \
    }
#else
#define GEMM_K_LOOP(ldsB_, acc_)                                                       \
    for (int kt = 0; kt < Dd / 32; ++kt) {                                             \
        __syncthreads();                                                               \
        *(uint4*)&(ldsB_)[0][tcol][tpart] = *(const uint4*)(gB + kt * 32);             \
        __syncthreads();                                                               \
        const int k0 = kt * 32;                                                        \
        if (k0 + 32 < Dd) __builtin_prefetch(arow + k0 + 32, 0, 3);                    \
        v16h a = load_frag_a(arow, k0, lane);                                          \
        _Pragma("unroll") for (int t = 0; t < 4; ++t) {                                \
            v16h b = load_frag_b(&(ldsB_)[0][t * 16 + nn][0], 0, lane);                \
            (acc_)[t] = WMMA_F32_F16(a, b, (acc_)[t]);                                 \
        }                                                                              \
    }
#endif

// ---------------- QKV projection GEMM ----------------
// vmode 0: out[((b*H+h)*S + s)*DH + dh]   (Q, K per-head row-major)
// vmode 1: out[((b*H+h)*DH + dh)*S + s]   (V per-head transposed, key-contiguous)
__global__ __launch_bounds__(256) void gemm_qkv(const _Float16* __restrict__ A,
                                                const _Float16* __restrict__ Bt,
                                                _Float16* __restrict__ out, int vmode) {
    __shared__ __align__(16) _Float16 ldsB[2][64][LDSP];
    v8f acc[4];
    GEMM_ACC_BODY(A, Bt, ldsB, acc)
    GEMM_K_LOOP(ldsB, acc)

    const int mrow = (lane < 16) ? 0 : 8;
#pragma unroll
    for (int t = 0; t < 4; ++t) {
#pragma unroll
        for (int r = 0; r < 8; ++r) {
            int m  = m0 + r + mrow;
            int n  = n0 + t * 16 + nn;
            int b  = m >> 12;          // m / S
            int s  = m & (Ss - 1);
            int h  = n >> 6;           // n / DH
            int dh = n & (DHh - 1);
            size_t idx = (vmode == 0)
                ? ((size_t)(b * Hh + h) * Ss + s) * DHh + dh
                : ((size_t)(b * Hh + h) * DHh + dh) * Ss + s;
            out[idx] = (_Float16)acc[t][r];
        }
    }
}

// ---------------- output projection GEMM (f32 out + bias) ----------------
__global__ __launch_bounds__(256) void gemm_out(const _Float16* __restrict__ A,
                                                const _Float16* __restrict__ Bt,
                                                const float* __restrict__ bias,
                                                float* __restrict__ out) {
    __shared__ __align__(16) _Float16 ldsB[2][64][LDSP];
    v8f acc[4];
    GEMM_ACC_BODY(A, Bt, ldsB, acc)
    GEMM_K_LOOP(ldsB, acc)

    const int mrow = (lane < 16) ? 0 : 8;
#pragma unroll
    for (int t = 0; t < 4; ++t) {
        int n = n0 + t * 16 + nn;
        float bv = bias[n];
#pragma unroll
        for (int r = 0; r < 8; ++r) {
            int m = m0 + r + mrow;
            out[(size_t)m * Dd + n] = acc[t][r] + bv;
        }
    }
}

// ---------------- causal flash attention ----------------
// Qh,Kh: [B*H][S][DH] f16; Vt: [B*H][DH][S] f16; ctx: [B][S][D] f16 (D = H*DH)
__global__ __launch_bounds__(128) void attn_causal(const _Float16* __restrict__ Qh,
                                                   const _Float16* __restrict__ Kh,
                                                   const _Float16* __restrict__ Vt,
                                                   _Float16* __restrict__ ctx) {
    __shared__ __align__(32) _Float16 plds[4][16][32];   // per-wave P staging (f16)

    const int lane  = threadIdx.x & 31;
    const int w     = threadIdx.x >> 5;
    const int qbase = (blockIdx.x * 4 + w) * 16;         // gridDim.x = S/64
    const int bh    = blockIdx.y;                        // gridDim.y = B*H
    const int b     = bh >> 4;
    const int h     = bh & (Hh - 1);

    const _Float16* Q = Qh + (size_t)bh * Ss * DHh;
    const _Float16* K = Kh + (size_t)bh * Ss * DHh;
    const _Float16* V = Vt + (size_t)bh * DHh * Ss;

    // Q fragments (16 queries x 64 dh = 2 K-chunks of 32)
    const _Float16* qrow = Q + (size_t)(qbase + (lane & 15)) * DHh;
    v16h qa0 = load_frag_a(qrow, 0, lane);
    v16h qa1 = load_frag_a(qrow, 32, lane);

    const v8f zero = {0.f, 0.f, 0.f, 0.f, 0.f, 0.f, 0.f, 0.f};
    v8f acc[4];
    float rowM[8], rowL[8];
#pragma unroll
    for (int t = 0; t < 4; ++t) acc[t] = zero;
#pragma unroll
    for (int r = 0; r < 8; ++r) { rowM[r] = -__builtin_huge_valf(); rowL[r] = 0.f; }

    const float scale = 0.125f;   // 1/sqrt(DH)
    const int   mrow  = (lane < 16) ? 0 : 8;
    const int   nn    = lane & 15;

    for (int jb = 0; jb <= qbase; jb += 32) {
        // scores: two 16x16 tiles (keys jb..jb+15, jb+16..jb+31), K-dim = DH = 64
        const _Float16* kc0 = K + (size_t)(jb + nn) * DHh;
        const _Float16* kc1 = K + (size_t)(jb + 16 + nn) * DHh;
        v8f s0 = zero, s1 = zero;
        s0 = WMMA_F32_F16(qa0, load_frag_b(kc0, 0, lane), s0);
        s0 = WMMA_F32_F16(qa1, load_frag_b(kc0, 32, lane), s0);
        s1 = WMMA_F32_F16(qa0, load_frag_b(kc1, 0, lane), s1);
        s1 = WMMA_F32_F16(qa1, load_frag_b(kc1, 32, lane), s1);

        // online softmax per row (rows replicated across the 16-lane halves)
#pragma unroll
        for (int r = 0; r < 8; ++r) {
            int qg  = qbase + r + mrow;
            int kg0 = jb + nn;
            float f0 = (kg0      <= qg) ? s0[r] * scale : -__builtin_huge_valf();
            float f1 = (kg0 + 16 <= qg) ? s1[r] * scale : -__builtin_huge_valf();

            float bm = fmaxf(f0, f1);
            bm = fmaxf(bm, __shfl_xor(bm, 1));
            bm = fmaxf(bm, __shfl_xor(bm, 2));
            bm = fmaxf(bm, __shfl_xor(bm, 4));
            bm = fmaxf(bm, __shfl_xor(bm, 8));

            float newM = fmaxf(rowM[r], bm);
            float fac  = __expf(rowM[r] - newM);
            rowM[r] = newM;

            float p0 = __expf(f0 - newM);
            float p1 = __expf(f1 - newM);
            float rs = p0 + p1;
            rs += __shfl_xor(rs, 1);
            rs += __shfl_xor(rs, 2);
            rs += __shfl_xor(rs, 4);
            rs += __shfl_xor(rs, 8);
            rowL[r] = rowL[r] * fac + rs;

#pragma unroll
            for (int t = 0; t < 4; ++t) acc[t][r] *= fac;

            int m = r + mrow;
            plds[w][m][nn]      = (_Float16)p0;
            plds[w][m][nn + 16] = (_Float16)p1;
        }

        // drain LDS stores before re-reading P in A-fragment layout (same wave only)
        asm volatile("s_wait_dscnt 0" ::: "memory");

        v16h pa = load_frag_a(&plds[w][lane & 15][0], 0, lane);

        // ctx(16x64) += P(16x32) x V(32x64): 4 N-tiles, K-dim = 32 keys
#pragma unroll
        for (int t = 0; t < 4; ++t) {
            const _Float16* vcol = V + (size_t)(t * 16 + nn) * Ss + jb;
            v16h vb = load_frag_b(vcol, 0, lane);
            acc[t] = WMMA_F32_F16(pa, vb, acc[t]);
        }
    }

    // normalize and store ctx as [B][S][H*DH] f16
#pragma unroll
    for (int r = 0; r < 8; ++r) {
        float inv = 1.0f / rowL[r];
        int s = qbase + r + mrow;
#pragma unroll
        for (int t = 0; t < 4; ++t) {
            int col = h * DHh + t * 16 + nn;
            ctx[((size_t)b * Ss + s) * Dd + col] = (_Float16)(acc[t][r] * inv);
        }
    }
}

// ---------------- host launcher ----------------

extern "C" void kernel_launch(void* const* d_in, const int* in_sizes, int n_in,
                              void* d_out, int out_size, void* d_ws, size_t ws_size,
                              hipStream_t stream) {
    const float* x  = (const float*)d_in[0];
    const float* Wq = (const float*)d_in[1];
    const float* Wk = (const float*)d_in[2];
    const float* Wv = (const float*)d_in[3];
    const float* Wo = (const float*)d_in[4];
    const float* bo = (const float*)d_in[5];
    float* out = (float*)d_out;

    char* ws = (char*)d_ws;
    size_t off = 0;
    auto carve = [&](size_t bytes) { char* p = ws + off; off += (bytes + 255) & ~(size_t)255; return p; };

    _Float16* x16  = (_Float16*)carve((size_t)BS * Dd * 2);
    _Float16* WqT  = (_Float16*)carve((size_t)Dd * Dd * 2);
    _Float16* WkT  = (_Float16*)carve((size_t)Dd * Dd * 2);
    _Float16* WvT  = (_Float16*)carve((size_t)Dd * Dd * 2);
    _Float16* WoT  = (_Float16*)carve((size_t)Dd * Dd * 2);
    _Float16* Qh   = (_Float16*)carve((size_t)Bb * Hh * Ss * DHh * 2);
    _Float16* Kh   = (_Float16*)carve((size_t)Bb * Hh * Ss * DHh * 2);
    _Float16* Vt   = (_Float16*)carve((size_t)Bb * Hh * Ss * DHh * 2);
    _Float16* ctx  = (_Float16*)carve((size_t)BS * Dd * 2);

    // 1) precision conversion + weight transposes
    {
        int n = BS * Dd;
        cvt_f32_to_f16<<<(n + 255) / 256, 256, 0, stream>>>(x, x16, n);
        int nw = Dd * Dd;
        transpose_w_f16<<<(nw + 255) / 256, 256, 0, stream>>>(Wq, WqT);
        transpose_w_f16<<<(nw + 255) / 256, 256, 0, stream>>>(Wk, WkT);
        transpose_w_f16<<<(nw + 255) / 256, 256, 0, stream>>>(Wv, WvT);
        transpose_w_f16<<<(nw + 255) / 256, 256, 0, stream>>>(Wo, WoT);
    }

    // 2) QKV projections
    dim3 ggrid(Dd / 64, BS / 128);
    gemm_qkv<<<ggrid, 256, 0, stream>>>(x16, WqT, Qh, 0);
    gemm_qkv<<<ggrid, 256, 0, stream>>>(x16, WkT, Kh, 0);
    gemm_qkv<<<ggrid, 256, 0, stream>>>(x16, WvT, Vt, 1);

    // 3) causal attention
    attn_causal<<<dim3(Ss / 64, Bb * Hh), 128, 0, stream>>>(Qh, Kh, Vt, ctx);

    // 4) output projection + bias (f32 out)
    gemm_out<<<ggrid, 256, 0, stream>>>(ctx, WoT, bo, out);
}
